// MoE_61100204753334
// MI455X (gfx1250) — matrix operand order repaired and checked by reading, not compiled, not run on hardware
//
#include <hip/hip_runtime.h>
#include <hip/hip_bf16.h>

#define NUM_EXPERTS 8
#define IN_DIM      2048
#define OUT_DIM     2048
#define BM          128
#define BN          128
#define BK          64

typedef __attribute__((ext_vector_type(4)))  _Float16 v4h;
typedef __attribute__((ext_vector_type(8)))  _Float16 v8h;
typedef __attribute__((ext_vector_type(16))) _Float16 v16h;
typedef __attribute__((ext_vector_type(4)))  float    v4f;
typedef __attribute__((ext_vector_type(8)))  float    v8f;

__device__ __forceinline__ unsigned lds_off32(const void* p) {
  // flat pointers to LDS carry the wave-relative LDS byte offset in addr[31:0]
  return (unsigned)(unsigned long long)p;
}

// ---------------------------------------------------------------- gating ----
__global__ void moe_gate_kernel(const float* __restrict__ x,
                                const float* __restrict__ gw,
                                const float* __restrict__ gb,
                                int*   __restrict__ top_idx,
                                float* __restrict__ top_p,
                                int*   __restrict__ counts, int B) {
  const int wid   = threadIdx.x >> 5;
  const int lane  = threadIdx.x & 31;
  const int token = blockIdx.x * (blockDim.x >> 5) + wid;
  if (token >= B) return;
  const float* xr = x + (size_t)token * IN_DIM;

  float acc[NUM_EXPERTS];
#pragma unroll
  for (int e = 0; e < NUM_EXPERTS; ++e) acc[e] = 0.f;
  for (int k = lane; k < IN_DIM; k += 32) {
    float xv = xr[k];
#pragma unroll
    for (int e = 0; e < NUM_EXPERTS; ++e) acc[e] += xv * gw[e * IN_DIM + k];
  }
#pragma unroll
  for (int e = 0; e < NUM_EXPERTS; ++e) {
    float v = acc[e];
#pragma unroll
    for (int off = 16; off > 0; off >>= 1) v += __shfl_xor(v, off, 32);
    acc[e] = v;
  }
  if (lane == 0) {
    float p[NUM_EXPERTS];
    float mx = -3.0e38f;
#pragma unroll
    for (int e = 0; e < NUM_EXPERTS; ++e) { p[e] = acc[e] + gb[e]; mx = fmaxf(mx, p[e]); }
    float sum = 0.f;
#pragma unroll
    for (int e = 0; e < NUM_EXPERTS; ++e) { p[e] = __expf(p[e] - mx); sum += p[e]; }
    float inv = 1.f / sum;
#pragma unroll
    for (int e = 0; e < NUM_EXPERTS; ++e) p[e] *= inv;
    int i0 = 0;
#pragma unroll
    for (int e = 1; e < NUM_EXPERTS; ++e) if (p[e] > p[i0]) i0 = e;
    int i1 = (i0 == 0) ? 1 : 0;
#pragma unroll
    for (int e = 0; e < NUM_EXPERTS; ++e) if (e != i0 && p[e] > p[i1]) i1 = e;
    top_idx[2 * token]     = i0;
    top_idx[2 * token + 1] = i1;
    top_p[2 * token]       = p[i0];
    top_p[2 * token + 1]   = p[i1];
    atomicAdd(&counts[i0], 1);
    atomicAdd(&counts[i1], 1);
  }
}

// ------------------------------------------------------------- exclusive scan
__global__ void moe_offsets_kernel(const int* __restrict__ counts,
                                   int* __restrict__ offsets) {
  if (threadIdx.x == 0) {
    int acc = 0;
    for (int e = 0; e < NUM_EXPERTS; ++e) { offsets[e] = acc; acc += counts[e]; }
  }
}

// ------------------------------------------------------------------ scatter -
__global__ void moe_scatter_kernel(const int* __restrict__ top_idx,
                                   const float* __restrict__ top_p,
                                   const int* __restrict__ offsets,
                                   int*   __restrict__ cursors,
                                   int*   __restrict__ bucket_tok,
                                   float* __restrict__ bucket_p, int B) {
  int t = blockIdx.x * blockDim.x + threadIdx.x;
  if (t >= B) return;
#pragma unroll
  for (int s = 0; s < 2; ++s) {
    int e   = top_idx[2 * t + s];
    int pos = atomicAdd(&cursors[e], 1);
    bucket_tok[offsets[e] + pos] = t;
    bucket_p[offsets[e] + pos]   = top_p[2 * t + s];
  }
}

// -------------------------------------------- out = p0*b[e0] + p1*b[e1] ----
__global__ void moe_init_out_kernel(const int* __restrict__ top_idx,
                                    const float* __restrict__ top_p,
                                    const float* __restrict__ eb,
                                    float* __restrict__ out) {
  int t  = blockIdx.x;
  int e0 = top_idx[2 * t], e1 = top_idx[2 * t + 1];
  float p0 = top_p[2 * t], p1 = top_p[2 * t + 1];
  const float* b0 = eb + (size_t)e0 * OUT_DIM;
  const float* b1 = eb + (size_t)e1 * OUT_DIM;
  for (int n = threadIdx.x; n < OUT_DIM; n += blockDim.x)
    out[(size_t)t * OUT_DIM + n] = p0 * b0[n] + p1 * b1[n];
}

// ---------------------------------------------------- f32 -> f16 pre-pass ---
__global__ void moe_cvt_f16_kernel(const float* __restrict__ src,
                                   _Float16* __restrict__ dst, long n) {
  long i = ((long)blockIdx.x * blockDim.x + threadIdx.x) * 8;
  if (i >= n) return;
  v4f a = *(const v4f*)(src + i);
  v4f b = *(const v4f*)(src + i + 4);
  v8h h;
  h[0] = (_Float16)a[0]; h[1] = (_Float16)a[1]; h[2] = (_Float16)a[2]; h[3] = (_Float16)a[3];
  h[4] = (_Float16)b[0]; h[5] = (_Float16)b[1]; h[6] = (_Float16)b[2]; h[7] = (_Float16)b[3];
  *(v8h*)(dst + i) = h;
}

// ======================= f16 grouped GEMM, async-LDS double-buffered ========
__global__ __launch_bounds__(256)
void moe_gemm_f16_kernel(const _Float16* __restrict__ x16,
                         const _Float16* __restrict__ w16,
                         const int*   __restrict__ counts,
                         const int*   __restrict__ offsets,
                         const int*   __restrict__ bucket_tok,
                         const float* __restrict__ bucket_p,
                         float* __restrict__ out) {
  const int e   = blockIdx.z;
  const int cnt = counts[e];
  const int m0  = blockIdx.x * BM;
  if (m0 >= cnt) return;
  const int off = offsets[e];
  const int n0  = blockIdx.y * BN;

  __shared__ __align__(16) _Float16 As[2][BM][BK + 8];
  __shared__ __align__(16) _Float16 Bs[2][BN][BK + 8];
  __shared__ int   s_tok[BM];
  __shared__ float s_prob[BM];

  const int tid  = threadIdx.x;
  const int wid  = tid >> 5;
  const int lane = tid & 31;
  const int wm   = (wid >> 2) * 64;
  const int wn   = (wid & 3) * 32;
  const int hl   = lane >> 4;
  const int l15  = lane & 15;

  if (tid < BM) {
    int pos = m0 + tid;
    if (pos < cnt) { s_tok[tid] = bucket_tok[off + pos]; s_prob[tid] = bucket_p[off + pos]; }
    else           { s_tok[tid] = -1;          s_prob[tid] = 0.f; }
  }
  __syncthreads();                       // s_tok ready before first fill

  // per-thread tile-fill assignment: 2 threads/row, 64B (4 x b128) each
  const int frow = tid >> 1;
  const int fcol = (tid & 1) * 32;       // halves
  const int ftok = s_tok[frow];
  const unsigned a_voff_base =
      (ftok >= 0) ? (unsigned)(((unsigned)ftok * IN_DIM) * 2 + fcol * 2) : 0u;
  const unsigned b_voff_base =
      (unsigned)((((unsigned)e * OUT_DIM + n0 + frow) * IN_DIM) * 2 + fcol * 2);

  auto fill = [&](int buf, int k0) {
    unsigned albase = lds_off32(&As[buf][frow][fcol]);
    unsigned blbase = lds_off32(&Bs[buf][frow][fcol]);
    if (ftok >= 0) {
#pragma unroll
      for (int j = 0; j < 4; ++j)
        asm volatile("global_load_async_to_lds_b128 %0, %1, %2"
                     :: "v"(albase + j * 16),
                        "v"(a_voff_base + (unsigned)k0 * 2 + j * 16),
                        "s"(x16)
                     : "memory");
    } else {
      v8h z = {};
#pragma unroll
      for (int j = 0; j < 4; ++j)
        *(v8h*)&As[buf][frow][fcol + j * 8] = z;
    }
#pragma unroll
    for (int j = 0; j < 4; ++j)
      asm volatile("global_load_async_to_lds_b128 %0, %1, %2"
                   :: "v"(blbase + j * 16),
                      "v"(b_voff_base + (unsigned)k0 * 2 + j * 16),
                      "s"(w16)
                   : "memory");
  };

  fill(0, 0);                            // prologue

  v8f acc[4][2] = {};
  const int ITERS = IN_DIM / BK;

  for (int it = 0; it < ITERS; ++it) {
    const int cur = it & 1;
    asm volatile("s_wait_asynccnt 0x0" ::: "memory");  // this wave's fills done
    __syncthreads();                                   // all waves' fills done
    if (it + 1 < ITERS) fill(cur ^ 1, (it + 1) * BK);  // overlap next tile

#pragma unroll
    for (int kk = 0; kk < BK; kk += 32) {
      v16h a[4], b[2];
#pragma unroll
      for (int mf = 0; mf < 4; ++mf) {
        const _Float16* p = &As[cur][wm + mf * 16 + l15][kk + (hl << 3)];
        v8h lo = *(const v8h*)p;
        v8h hi = *(const v8h*)(p + 16);
        a[mf] = __builtin_shufflevector(lo, hi, 0, 1, 2, 3, 4, 5, 6, 7,
                                                8, 9, 10, 11, 12, 13, 14, 15);
      }
#pragma unroll
      for (int nf = 0; nf < 2; ++nf) {
        const _Float16* p = &Bs[cur][wn + nf * 16 + l15][kk + (hl << 3)];
        v8h lo = *(const v8h*)p;
        v8h hi = *(const v8h*)(p + 16);
        b[nf] = __builtin_shufflevector(lo, hi, 0, 1, 2, 3, 4, 5, 6, 7,
                                                8, 9, 10, 11, 12, 13, 14, 15);
      }
#pragma unroll
      for (int mf = 0; mf < 4; ++mf)
#pragma unroll
        for (int nf = 0; nf < 2; ++nf)
          acc[mf][nf] = __builtin_amdgcn_wmma_f32_16x16x32_f16(
              false, a[mf], false, b[nf], (short)0, acc[mf][nf], false, false);
    }
  }

#pragma unroll
  for (int mf = 0; mf < 4; ++mf)
#pragma unroll
    for (int nf = 0; nf < 2; ++nf) {
      size_t ncol = (size_t)n0 + wn + nf * 16 + l15;
#pragma unroll
      for (int i = 0; i < 8; ++i) {
        int mlocal = wm + mf * 16 + (hl << 3) + i;
        int tok = s_tok[mlocal];
        if (tok >= 0)
          atomicAdd(&out[(size_t)tok * OUT_DIM + ncol],
                    s_prob[mlocal] * acc[mf][nf][i]);
      }
    }
}

// ======================= f32 fallback GEMM (small workspace) ================
__global__ __launch_bounds__(256)
void moe_gemm_f32_kernel(const float* __restrict__ x,
                         const float* __restrict__ ew,
                         const int*   __restrict__ counts,
                         const int*   __restrict__ offsets,
                         const int*   __restrict__ bucket_tok,
                         const float* __restrict__ bucket_p,
                         float* __restrict__ out) {
  const int e   = blockIdx.z;
  const int cnt = counts[e];
  const int m0  = blockIdx.x * BM;
  if (m0 >= cnt) return;
  const int off = offsets[e];
  const int n0  = blockIdx.y * BN;

  __shared__ __align__(16) _Float16 As[BM][BK + 8];
  __shared__ __align__(16) _Float16 Bs[BN][BK + 8];
  __shared__ int   s_tok[BM];
  __shared__ float s_prob[BM];

  const int tid  = threadIdx.x;
  const int wid  = tid >> 5;
  const int lane = tid & 31;
  const int wm   = (wid >> 2) * 64;
  const int wn   = (wid & 3) * 32;
  const int hl   = lane >> 4;
  const int l15  = lane & 15;

  if (tid < BM) {
    int pos = m0 + tid;
    if (pos < cnt) { s_tok[tid] = bucket_tok[off + pos]; s_prob[tid] = bucket_p[off + pos]; }
    else           { s_tok[tid] = -1;          s_prob[tid] = 0.f; }
  }

  v8f acc[4][2] = {};
  const int lrow = tid >> 4;
  const int lcol = (tid & 15) << 2;

  for (int k0 = 0; k0 < IN_DIM; k0 += BK) {
    __syncthreads();
#pragma unroll
    for (int pass = 0; pass < 8; ++pass) {
      int row = pass * 16 + lrow;
      int tok = s_tok[row];
      v4f av = {0.f, 0.f, 0.f, 0.f};
      if (tok >= 0) av = *(const v4f*)(x + (size_t)tok * IN_DIM + k0 + lcol);
      v4h ah; ah[0] = (_Float16)av[0]; ah[1] = (_Float16)av[1];
              ah[2] = (_Float16)av[2]; ah[3] = (_Float16)av[3];
      *(v4h*)&As[row][lcol] = ah;
      v4f bv = *(const v4f*)(ew + (size_t)(e * OUT_DIM + n0 + row) * IN_DIM + k0 + lcol);
      v4h bh; bh[0] = (_Float16)bv[0]; bh[1] = (_Float16)bv[1];
              bh[2] = (_Float16)bv[2]; bh[3] = (_Float16)bv[3];
      *(v4h*)&Bs[row][lcol] = bh;
    }
    __syncthreads();

#pragma unroll
    for (int kk = 0; kk < BK; kk += 32) {
      v16h a[4], b[2];
#pragma unroll
      for (int mf = 0; mf < 4; ++mf) {
        const _Float16* p = &As[wm + mf * 16 + l15][kk + (hl << 3)];
        v8h lo = *(const v8h*)p;
        v8h hi = *(const v8h*)(p + 16);
        a[mf] = __builtin_shufflevector(lo, hi, 0, 1, 2, 3, 4, 5, 6, 7,
                                                8, 9, 10, 11, 12, 13, 14, 15);
      }
#pragma unroll
      for (int nf = 0; nf < 2; ++nf) {
        const _Float16* p = &Bs[wn + nf * 16 + l15][kk + (hl << 3)];
        v8h lo = *(const v8h*)p;
        v8h hi = *(const v8h*)(p + 16);
        b[nf] = __builtin_shufflevector(lo, hi, 0, 1, 2, 3, 4, 5, 6, 7,
                                                8, 9, 10, 11, 12, 13, 14, 15);
      }
#pragma unroll
      for (int mf = 0; mf < 4; ++mf)
#pragma unroll
        for (int nf = 0; nf < 2; ++nf)
          acc[mf][nf] = __builtin_amdgcn_wmma_f32_16x16x32_f16(
              false, a[mf], false, b[nf], (short)0, acc[mf][nf], false, false);
    }
  }

#pragma unroll
  for (int mf = 0; mf < 4; ++mf)
#pragma unroll
    for (int nf = 0; nf < 2; ++nf) {
      size_t ncol = (size_t)n0 + wn + nf * 16 + l15;
#pragma unroll
      for (int i = 0; i < 8; ++i) {
        int mlocal = wm + mf * 16 + (hl << 3) + i;
        int tok = s_tok[mlocal];
        if (tok >= 0)
          atomicAdd(&out[(size_t)tok * OUT_DIM + ncol],
                    s_prob[mlocal] * acc[mf][nf][i]);
      }
    }
}

// ---------------------------------------------------------------- launcher --
extern "C" void kernel_launch(void* const* d_in, const int* in_sizes, int n_in,
                              void* d_out, int out_size, void* d_ws, size_t ws_size,
                              hipStream_t stream) {
  const float* x  = (const float*)d_in[0];
  const float* ew = (const float*)d_in[1];
  const float* eb = (const float*)d_in[2];
  const float* gw = (const float*)d_in[3];
  const float* gb = (const float*)d_in[4];
  float* out = (float*)d_out;
  const int B = in_sizes[0] / IN_DIM;

  char* ws = (char*)d_ws;
  int*   counts     = (int*)(ws + 0);
  int*   offsets    = (int*)(ws + 32);
  int*   cursors    = (int*)(ws + 64);
  int*   top_idx    = (int*)(ws + 128);
  float* top_p      = (float*)(ws + 128 + (size_t)8 * B);
  int*   bucket_tok = (int*)(ws + 128 + (size_t)16 * B);
  float* bucket_p   = (float*)(ws + 128 + (size_t)24 * B);

  const size_t route_end = 128 + (size_t)32 * B;
  const size_t x16_off   = (route_end + 4095) & ~(size_t)4095;
  const size_t x16_bytes = (size_t)B * IN_DIM * 2;
  const size_t w16_off   = x16_off + x16_bytes;
  const size_t w16_bytes = (size_t)NUM_EXPERTS * OUT_DIM * IN_DIM * 2;
  const bool   use_f16   = ws_size >= (w16_off + w16_bytes);

  hipMemsetAsync(ws, 0, 128, stream);

  moe_gate_kernel<<<(B + 7) / 8, 256, 0, stream>>>(x, gw, gb, top_idx, top_p, counts, B);
  moe_offsets_kernel<<<1, 32, 0, stream>>>(counts, offsets);
  moe_scatter_kernel<<<(B + 255) / 256, 256, 0, stream>>>(top_idx, top_p, offsets,
                                                          cursors, bucket_tok, bucket_p, B);
  moe_init_out_kernel<<<B, 256, 0, stream>>>(top_idx, top_p, eb, out);

  dim3 grid((B + BM - 1) / BM, OUT_DIM / BN, NUM_EXPERTS);
  if (use_f16) {
    _Float16* x16 = (_Float16*)(ws + x16_off);
    _Float16* w16 = (_Float16*)(ws + w16_off);
    long nx = (long)B * IN_DIM;
    long nw = (long)NUM_EXPERTS * OUT_DIM * IN_DIM;
    moe_cvt_f16_kernel<<<(unsigned)(nx / (256 * 8)), 256, 0, stream>>>(x, x16, nx);
    moe_cvt_f16_kernel<<<(unsigned)(nw / (256 * 8)), 256, 0, stream>>>(ew, w16, nw);
    moe_gemm_f16_kernel<<<grid, 256, 0, stream>>>(x16, w16, counts, offsets,
                                                  bucket_tok, bucket_p, out);
  } else {
    moe_gemm_f32_kernel<<<grid, 256, 0, stream>>>(x, ew, counts, offsets,
                                                  bucket_tok, bucket_p, out);
  }
}
